// PointNetFeaturePropagation_60017872994425
// MI455X (gfx1250) — matrix operand order repaired and checked by reading, not compiled, not run on hardware
//
#include <hip/hip_runtime.h>
#include <hip/hip_bf16.h>

// Problem constants (match reference setup_inputs)
#define BB    8
#define NN    8192
#define SSZ   2048
#define D1C   64
#define D2C   256
#define CIN   320
#define COUT  256
#define MROWS (BB * NN)   // 65536

typedef __attribute__((ext_vector_type(16))) __bf16 v16bf;
typedef __attribute__((ext_vector_type(8)))  __bf16 v8bf;
typedef __attribute__((ext_vector_type(8)))  float  v8f;

__device__ __forceinline__ unsigned short f2bf(float f) {
  union { float f; unsigned u; } v; v.f = f;
  unsigned r = v.u + 0x7FFFu + ((v.u >> 16) & 1u);
  return (unsigned short)(r >> 16);
}

// Async global->LDS 128-bit copy (gfx1250). VDST = per-lane LDS byte offset
// (low 32 bits of the LDS-aperture flat address), VADDR = 64-bit global addr.
// Tracked by ASYNCcnt.
__device__ __forceinline__ void async_ld_b128(const void* g, void* lds) {
  const unsigned off = (unsigned)(size_t)lds;   // LDS byte offset = low 32 bits of pointer
  asm volatile("global_load_async_to_lds_b128 %0, %1, off"
               :: "v"(off), "v"(g) : "memory");
}
#define S_WAIT_ASYNC(n) asm volatile("s_wait_asynccnt %0" :: "n"(n) : "memory")

// ---------------------------------------------------------------- fp32 -> bf16
__global__ void cvt_bf16_kernel(const float* __restrict__ src,
                                unsigned short* __restrict__ dst, int n) {
  int i = blockIdx.x * 256 + threadIdx.x;
  if (i < n) dst[i] = f2bf(src[i]);
}

// ------------------------------------------------- [B,C,L] -> [B,L,C] (fp32)
__global__ void transpose_cl_kernel(const float* __restrict__ src,
                                    float* __restrict__ dst, int C, int L) {
  __shared__ float tile[32][33];
  const int l0 = blockIdx.x * 32, c0 = blockIdx.y * 32, b = blockIdx.z;
  const int tx = threadIdx.x & 31, ty = threadIdx.x >> 5;  // 32x8
  #pragma unroll
  for (int i = ty; i < 32; i += 8)
    tile[i][tx] = src[((size_t)b * C + c0 + i) * L + l0 + tx];
  __syncthreads();
  #pragma unroll
  for (int i = ty; i < 32; i += 8)
    dst[((size_t)b * L + l0 + i) * C + c0 + tx] = tile[tx][i];
}

// ----------------------------------- 3-NN + inverse-distance weights per query
__global__ void __launch_bounds__(256)
knn3_kernel(const float* __restrict__ xyz1, const float* __restrict__ xyz2,
            int* __restrict__ idx3, float* __restrict__ w3) {
  __shared__ float sx[SSZ], sy[SSZ], sz[SSZ];
  const int b    = blockIdx.x >> 5;          // 32 blocks of 256 queries / batch
  const int nblk = blockIdx.x & 31;
  const int n    = nblk * 256 + threadIdx.x;
  const float* p2 = xyz2 + (size_t)b * 3 * SSZ;
  for (int s = threadIdx.x; s < SSZ; s += 256) {
    sx[s] = p2[s]; sy[s] = p2[SSZ + s]; sz[s] = p2[2 * SSZ + s];
  }
  __syncthreads();
  const float* p1 = xyz1 + (size_t)b * 3 * NN;
  const float qx = p1[n], qy = p1[NN + n], qz = p1[2 * NN + n];
  float d0 = 1e30f, d1 = 1e30f, d2 = 1e30f;
  int   i0 = 0,     i1 = 0,     i2 = 0;
  for (int s = 0; s < SSZ; ++s) {
    const float dx = qx - sx[s], dy = qy - sy[s], dz = qz - sz[s];
    const float d = dx * dx + dy * dy + dz * dz;
    if (d < d2) {
      if (d < d1) {
        if (d < d0) { d2 = d1; i2 = i1; d1 = d0; i1 = i0; d0 = d; i0 = s; }
        else        { d2 = d1; i2 = i1; d1 = d;  i1 = s; }
      } else        { d2 = d;  i2 = s; }
    }
  }
  float w0 = 1.0f / (d0 + 1e-8f);
  float w1 = 1.0f / (d1 + 1e-8f);
  float w2 = 1.0f / (d2 + 1e-8f);
  const float inv = 1.0f / (w0 + w1 + w2);
  const size_t r = ((size_t)b * NN + n) * 3;
  idx3[r + 0] = i0; idx3[r + 1] = i1; idx3[r + 2] = i2;
  w3[r + 0] = w0 * inv; w3[r + 1] = w1 * inv; w3[r + 2] = w2 * inv;
}

// --------------- build concat row X0[row][0:320] = [points1 | interp] in bf16
__global__ void __launch_bounds__(256)
build_x0_kernel(const float* __restrict__ p1t, const float* __restrict__ f2t,
                const int* __restrict__ idx3, const float* __restrict__ w3,
                unsigned short* __restrict__ X0) {
  const int row = blockIdx.x;
  const int b   = row >> 13;  // /NN
  const int t   = threadIdx.x;
  __shared__ int   si[3];
  __shared__ float sw[3];
  if (t < 3) { si[t] = idx3[(size_t)row * 3 + t]; sw[t] = w3[(size_t)row * 3 + t]; }
  __syncthreads();
  unsigned short* xr = X0 + (size_t)row * CIN;
  if (t < D1C) xr[t] = f2bf(p1t[(size_t)row * D1C + t]);
  const float* fb = f2t + (size_t)b * SSZ * D2C;
  const float v = sw[0] * fb[(size_t)si[0] * D2C + t]
                + sw[1] * fb[(size_t)si[1] * D2C + t]
                + sw[2] * fb[(size_t)si[2] * D2C + t];
  xr[D1C + t] = f2bf(v);
}

// --------------------------------------------------------- bf16 WMMA GEMM
// Y[M,256] = A[M,K](bf16, row-major) * W[256,K](bf16, row-major)^T, fp32 acc.
// Block tile 128(M) x 64(Ncol); 8 waves, each wave one 16x64 strip (4 accs).
// Double-buffered LDS filled by async global->LDS DMA (ASYNCcnt), so the next
// K-tile streams in while WMMA consumes the current one.
__global__ void __launch_bounds__(256)
gemm_bf16_wmma_kernel(const unsigned short* __restrict__ A,
                      const unsigned short* __restrict__ Wb,
                      float* __restrict__ Y, int K) {
  __shared__ __align__(16) unsigned short Alds[2][128 * 32];  // [buf][row][k]
  __shared__ __align__(16) unsigned short Blds[2][64 * 32];   // [buf][col][k]
  const int m0 = blockIdx.x * 128;
  const int c0 = blockIdx.y * 64;
  const int t  = threadIdx.x;
  const int wave = t >> 5, lane = t & 31;
  const int lrow = lane & 15;
  const int ka   = (lane >> 4) << 4;   // A-frag: K contiguous by 16
  const int kb   = (lane >> 4) << 4;   // B-frag: K contiguous by 16

  // Per-thread staging coordinates (128-bit chunks of the K=32 tiles)
  const int ra0 = t >> 2,         qa0 = (t & 3) * 8;          // A chunk id t
  const int ra1 = (t + 256) >> 2, qa1 = (t & 3) * 8;          // A chunk id t+256
  const int rb  = t >> 2,         qb  = (t & 3) * 8;          // B chunk id t

  // 3 async b128 issues per wave per K-step
  auto issue_tiles = [&](int bsel, int kkk) {
    async_ld_b128(&A[(size_t)(m0 + ra0) * K + kkk + qa0], &Alds[bsel][t * 8]);
    async_ld_b128(&A[(size_t)(m0 + ra1) * K + kkk + qa1], &Alds[bsel][(t + 256) * 8]);
    async_ld_b128(&Wb[(size_t)(c0 + rb) * K + kkk + qb],  &Blds[bsel][t * 8]);
  };

  v8f acc0 = {}, acc1 = {}, acc2 = {}, acc3 = {};

  issue_tiles(0, 0);               // prologue: stage first K-tile
  int buf = 0;
  for (int kk = 0; kk < K; kk += 32) {
    if (kk + 32 < K) {
      issue_tiles(buf ^ 1, kk + 32);   // overlap next tile with compute
      S_WAIT_ASYNC(3);                 // current tile landed (in-order retire)
    } else {
      S_WAIT_ASYNC(0);
    }
    __syncthreads();                   // current tile visible to all waves

    // A fragment: elems 0-15 = K[ka..ka+15] (matches B)
    const v8bf* ap = (const v8bf*)&Alds[buf][(wave * 16 + lrow) * 32 + ka];
    const v8bf a_lo = ap[0];
    const v8bf a_hi = ap[1];
    const v16bf af = __builtin_shufflevector(a_lo, a_hi,
        0, 1, 2, 3, 4, 5, 6, 7, 8, 9, 10, 11, 12, 13, 14, 15);
    // B fragments in contiguous K order (same as A): lanes 0-15 K=0..15, lanes 16-31 K=16..31
    const v16bf bf0 = *(const v16bf*)&Blds[buf][(0 * 16 + lrow) * 32 + kb];
    const v16bf bf1 = *(const v16bf*)&Blds[buf][(1 * 16 + lrow) * 32 + kb];
    const v16bf bf2 = *(const v16bf*)&Blds[buf][(2 * 16 + lrow) * 32 + kb];
    const v16bf bf3 = *(const v16bf*)&Blds[buf][(3 * 16 + lrow) * 32 + kb];

    acc0 = __builtin_amdgcn_wmma_f32_16x16x32_bf16(false, af, false, bf0,
                                                   (short)0, acc0, false, false);
    acc1 = __builtin_amdgcn_wmma_f32_16x16x32_bf16(false, af, false, bf1,
                                                   (short)0, acc1, false, false);
    acc2 = __builtin_amdgcn_wmma_f32_16x16x32_bf16(false, af, false, bf2,
                                                   (short)0, acc2, false, false);
    acc3 = __builtin_amdgcn_wmma_f32_16x16x32_bf16(false, af, false, bf3,
                                                   (short)0, acc3, false, false);
    __syncthreads();   // WAR: all waves consumed buf before it is refilled
    buf ^= 1;
  }

  // Epilogue: C/D layout — VGPR r: lanes 0-15 M=r, lanes 16-31 M=r+8; N=lane&15.
  // Conv bias omitted: training-mode BN subtracts the batch mean, so it cancels.
  const int cb = lane & 15;
  const int mofs = (lane >> 4) << 3;
  #pragma unroll
  for (int r = 0; r < 8; ++r) {
    const size_t m = (size_t)m0 + wave * 16 + r + mofs;
    Y[m * COUT + c0 +  0 + cb] = acc0[r];
    Y[m * COUT + c0 + 16 + cb] = acc1[r];
    Y[m * COUT + c0 + 32 + cb] = acc2[r];
    Y[m * COUT + c0 + 48 + cb] = acc3[r];
  }
}

// ------------------------------------------------------------- BN statistics
__global__ void zero_stats_kernel(float* __restrict__ sum,
                                  float* __restrict__ sumsq) {
  sum[threadIdx.x] = 0.0f; sumsq[threadIdx.x] = 0.0f;
}

__global__ void __launch_bounds__(256)
bn_reduce_kernel(const float* __restrict__ Y, float* __restrict__ sum,
                 float* __restrict__ sumsq) {
  const int t = threadIdx.x;                 // channel
  const size_t r0 = (size_t)blockIdx.x * 256;
  float s = 0.0f, q = 0.0f;
  for (int r = 0; r < 256; ++r) {
    const float v = Y[(r0 + r) * COUT + t];
    s += v; q += v * v;
  }
  atomicAdd(&sum[t], s);
  atomicAdd(&sumsq[t], q);
}

__global__ void bn_finalize_kernel(const float* __restrict__ sum,
                                   const float* __restrict__ sumsq,
                                   const float* __restrict__ gamma,
                                   const float* __restrict__ beta,
                                   float* __restrict__ scale,
                                   float* __restrict__ shift) {
  const int c = threadIdx.x;
  const float inv_m = 1.0f / (float)MROWS;
  const float mu  = sum[c] * inv_m;
  const float var = sumsq[c] * inv_m - mu * mu;
  const float sc  = gamma[c] * rsqrtf(var + 1e-5f);
  scale[c] = sc;
  shift[c] = beta[c] - mu * sc;
}

// ------------------------------------- BN+ReLU -> bf16 (input of next layer)
__global__ void __launch_bounds__(256)
bn_apply_bf16_kernel(const float* __restrict__ Y, const float* __restrict__ scale,
                     const float* __restrict__ shift,
                     unsigned short* __restrict__ X1) {
  const size_t i = (size_t)blockIdx.x * 256 + threadIdx.x;
  const int c = threadIdx.x;  // blockDim == COUT, row-major [., 256]
  const float v = Y[i] * scale[c] + shift[c];
  X1[i] = f2bf(fmaxf(v, 0.0f));
}

// ------------- final BN+ReLU + transpose [M,256] -> out [B,256,N] via LDS tile
__global__ void __launch_bounds__(256)
bn_apply_out_kernel(const float* __restrict__ Y, const float* __restrict__ scale,
                    const float* __restrict__ shift, float* __restrict__ out) {
  __shared__ float tile[32][65];             // [channel][n], padded
  const int n0 = blockIdx.x * 64, c0 = blockIdx.y * 32, b = blockIdx.z;
  const int t = threadIdx.x;
  const size_t m0 = (size_t)b * NN + n0;
  #pragma unroll
  for (int u = 0; u < 8; ++u) {
    const int id = u * 256 + t;
    const int r = id >> 5, c = id & 31;      // coalesced read of Y rows
    const float v = Y[(m0 + r) * COUT + c0 + c];
    tile[c][r] = fmaxf(v * scale[c0 + c] + shift[c0 + c], 0.0f);
  }
  __syncthreads();
  #pragma unroll
  for (int u = 0; u < 8; ++u) {
    const int id = u * 256 + t;
    const int c = id >> 6, n = id & 63;      // coalesced write of out rows
    out[((size_t)b * COUT + c0 + c) * NN + n0 + n] = tile[c][n];
  }
}

// =============================================================== launch glue
extern "C" void kernel_launch(void* const* d_in, const int* in_sizes, int n_in,
                              void* d_out, int out_size, void* d_ws, size_t ws_size,
                              hipStream_t stream) {
  (void)in_sizes; (void)n_in; (void)out_size; (void)ws_size;
  const float* xyz1   = (const float*)d_in[0];   // [B,3,N]
  const float* xyz2   = (const float*)d_in[1];   // [B,3,S]
  const float* pts1   = (const float*)d_in[2];   // [B,64,N]
  const float* pts2   = (const float*)d_in[3];   // [B,256,S]
  const float* W0     = (const float*)d_in[4];   // [256,320]
  const float* gamma0 = (const float*)d_in[6];
  const float* beta0  = (const float*)d_in[7];
  const float* W1     = (const float*)d_in[8];   // [256,256]
  const float* gamma1 = (const float*)d_in[10];
  const float* beta1  = (const float*)d_in[11];
  float* out = (float*)d_out;                    // [B,256,N]

  // Workspace layout (bias b0/b1 unused: cancels under training-mode BN)
  char* ws = (char*)d_ws;
  size_t o = 0;
  int*            idx3 = (int*)(ws + o);            o += (size_t)MROWS * 3 * 4;
  float*          w3   = (float*)(ws + o);          o += (size_t)MROWS * 3 * 4;
  char*           reg32 = ws + o;                   o += (size_t)32 * 1024 * 1024;
  float*          f2t  = (float*)reg32;                       // [B,S,256]  16MB
  float*          p1t  = (float*)(reg32 + 16u * 1024 * 1024); // [B,N,64]   16MB
  unsigned short* X1   = (unsigned short*)reg32;              // aliases (dead after X0)
  unsigned short* X0   = (unsigned short*)(ws + o); o += (size_t)MROWS * CIN * 2;
  unsigned short* Wb0  = (unsigned short*)(ws + o); o += (size_t)COUT * CIN * 2;
  unsigned short* Wb1  = (unsigned short*)(ws + o); o += (size_t)COUT * COUT * 2;
  float*          Yb   = (float*)(ws + o);          o += (size_t)MROWS * COUT * 4;
  float*          ssum  = (float*)(ws + o);         o += 1024;
  float*          ssq   = (float*)(ws + o);         o += 1024;
  float*          scale = (float*)(ws + o);         o += 1024;
  float*          shift = (float*)(ws + o);         o += 1024;

  // 1. Weights to bf16
  cvt_bf16_kernel<<<(COUT * CIN + 255) / 256, 256, 0, stream>>>(W0, Wb0, COUT * CIN);
  cvt_bf16_kernel<<<(COUT * COUT + 255) / 256, 256, 0, stream>>>(W1, Wb1, COUT * COUT);

  // 2. Feature transposes: points2 [B,256,S]->[B,S,256]; points1 [B,64,N]->[B,N,64]
  transpose_cl_kernel<<<dim3(SSZ / 32, D2C / 32, BB), 256, 0, stream>>>(pts2, f2t, D2C, SSZ);
  transpose_cl_kernel<<<dim3(NN / 32, D1C / 32, BB), 256, 0, stream>>>(pts1, p1t, D1C, NN);

  // 3. 3-NN + interpolation weights
  knn3_kernel<<<BB * (NN / 256), 256, 0, stream>>>(xyz1, xyz2, idx3, w3);

  // 4. Build concatenated bf16 activation X0 [M,320]
  build_x0_kernel<<<MROWS, 256, 0, stream>>>(p1t, f2t, idx3, w3, X0);

  // 5. Layer 0: GEMM (WMMA) -> BN stats -> BN+ReLU -> bf16 X1
  gemm_bf16_wmma_kernel<<<dim3(MROWS / 128, COUT / 64), 256, 0, stream>>>(X0, Wb0, Yb, CIN);
  zero_stats_kernel<<<1, 256, 0, stream>>>(ssum, ssq);
  bn_reduce_kernel<<<MROWS / 256, 256, 0, stream>>>(Yb, ssum, ssq);
  bn_finalize_kernel<<<1, 256, 0, stream>>>(ssum, ssq, gamma0, beta0, scale, shift);
  bn_apply_bf16_kernel<<<MROWS * (COUT / 256), 256, 0, stream>>>(Yb, scale, shift, X1);

  // 6. Layer 1: GEMM (WMMA) -> BN stats -> BN+ReLU + transpose to d_out
  gemm_bf16_wmma_kernel<<<dim3(MROWS / 128, COUT / 64), 256, 0, stream>>>(X1, Wb1, Yb, COUT);
  zero_stats_kernel<<<1, 256, 0, stream>>>(ssum, ssq);
  bn_reduce_kernel<<<MROWS / 256, 256, 0, stream>>>(Yb, ssum, ssq);
  bn_finalize_kernel<<<1, 256, 0, stream>>>(ssum, ssq, gamma1, beta1, scale, shift);
  bn_apply_out_kernel<<<dim3(NN / 64, COUT / 32, BB), 256, 0, stream>>>(Yb, scale, shift, out);
}
